// DeepSelfAttention_86346022518823
// MI455X (gfx1250) — matrix operand order repaired and checked by reading, not compiled, not run on hardware
//
#include <hip/hip_runtime.h>
#include <hip/hip_bf16.h>

// ---------------------------------------------------------------------------
// Types matching the probe-verified WMMA builtin signature:
//   __builtin_amdgcn_wmma_f32_16x16x32_bf16(bool, v16bf, bool, v16bf, short, v8f, bool, bool)
// ---------------------------------------------------------------------------
typedef __attribute__((ext_vector_type(16))) __bf16 v16bf;
typedef __attribute__((ext_vector_type(8)))  __bf16 v8bf;
typedef __attribute__((ext_vector_type(8)))  float  v8f;

#define LOG2E 1.44269504088896340736f
#define ATTN_SCALE 0.0625f   // 1/sqrt(256)

#if defined(__AMDGCN__) && __has_builtin(__builtin_amdgcn_global_load_async_to_lds_b128)
#define USE_ASYNC 1
#else
#define USE_ASYNC 0
#endif

__device__ __forceinline__ v8f wmma_bf16(v16bf a, v16bf b, v8f c) {
  return __builtin_amdgcn_wmma_f32_16x16x32_bf16(false, a, false, b, (short)0, c,
                                                 false, false);
}

#if USE_ASYNC
// Exact parameter type per hipcc diagnostic: pointer to int __vector_size__(16)
// in the respective address space (1 = global source, 3 = LDS destination).
typedef int v4i_vs __attribute__((vector_size(4 * sizeof(int))));
typedef __attribute__((address_space(1))) v4i_vs gv4i;
typedef __attribute__((address_space(3))) v4i_vs lv4i;

__device__ __forceinline__ void async_b128(const void* g, void* l) {
  // global_load_async_to_lds_b128 (tracked by ASYNCcnt); 16B per lane.
  __builtin_amdgcn_global_load_async_to_lds_b128((gv4i*)g, (lv4i*)l, 0, 0);
}
__device__ __forceinline__ void wait_async0() {
#if __has_builtin(__builtin_amdgcn_s_wait_asynccnt)
  __builtin_amdgcn_s_wait_asynccnt(0);
#else
  asm volatile("s_wait_asynccnt 0" ::: "memory");
#endif
}
#endif

// ---------------------------------------------------------------------------
// Fragment loaders. Caller precomputes the per-lane base pointer once; the
// constant k-offset then folds into the load's immediate offset field.
//   A (M=16,K=32) lane base: base + (row0 + (lane&15))*ld + ((lane>>4)&1)*16
//     Layout: lanes 0-15 hold K[0..16), lanes 16-31 hold K[16..32), contiguous.
//   B (K=32,N=16) lane base (B^T row-major): baseT + (col0+(lane&15))*ld + ((lane>>4)&1)*16
//     Layout (same as A, not the ISA table order): lanes 0-15 hold K[0..16), lanes 16-31 hold K[16..32), contiguous.
// ---------------------------------------------------------------------------
__device__ __forceinline__ v16bf a_frag_bf_at(const __bf16* laneBase, int k0) {
  v8bf lo  = *(const v8bf*)(laneBase + k0);
  v8bf hi8 = *(const v8bf*)(laneBase + k0 + 8);
  return __builtin_shufflevector(lo, hi8, 0, 1, 2, 3, 4, 5, 6, 7,
                                 8, 9, 10, 11, 12, 13, 14, 15);
}

__device__ __forceinline__ v16bf b_frag_bf_at(const __bf16* laneBase, int k0) {
  return *(const v16bf*)(laneBase + k0);
}

__device__ __forceinline__ v16bf a_frag_f32_at(const float* laneBase, int k0,
                                               float scale) {
  const float* p0 = laneBase + k0;
  const float* p1 = p0 + 8;
  v16bf r;
#pragma unroll
  for (int i = 0; i < 8; ++i) {
    r[i]     = (__bf16)(p0[i] * scale);
    r[i + 8] = (__bf16)(p1[i] * scale);
  }
  return r;
}

__device__ __forceinline__ v16bf b_frag_f32_at(const float* laneBase, int k0) {
  const float* p = laneBase + k0;
  v16bf r;
#pragma unroll
  for (int i = 0; i < 16; ++i) r[i] = (__bf16)p[i];
  return r;
}

// ===========================================================================
// Phase 1: per group of blocks, compute Q, K (row-major [4096,256] bf16) and
// Vt ([256,4096] bf16, V transposed) so attention only does contiguous
// fragment loads. C = A @ B^T + bias, bf16 out.
// grid = (128, 3, cnt), block = 256 (8 waves), wave tile = 16x64.
// ===========================================================================
__global__ __launch_bounds__(256) void qkv_gemm(
    const float* __restrict__ x, const float* __restrict__ Wq,
    const float* __restrict__ Wk, const float* __restrict__ Wv,
    const float* __restrict__ bq, const float* __restrict__ bk,
    const float* __restrict__ bv, __bf16* __restrict__ Qs,
    __bf16* __restrict__ Ks, __bf16* __restrict__ Vts, int b0) {
  int tid = threadIdx.x, lane = tid & 31, wid = tid >> 5;
  int which = blockIdx.y;
  int slot = blockIdx.z;
  int blk = b0 + slot;

  const float *A, *B, *bias;
  __bf16* C;
  int N;
  bool biasRow;
  if (which == 0) {
    A = x; B = Wq + (size_t)blk * 65536; bias = bq + (size_t)blk * 256;
    C = Qs + (size_t)slot * 1048576; N = 256; biasRow = false;
  } else if (which == 1) {
    A = x; B = Wk + (size_t)blk * 65536; bias = bk + (size_t)blk * 256;
    C = Ks + (size_t)slot * 1048576; N = 256; biasRow = false;
  } else {  // Vt = Wv @ x^T + bv (bias indexed by output row)
    A = Wv + (size_t)blk * 65536; B = x; bias = bv + (size_t)blk * 256;
    C = Vts + (size_t)slot * 1048576; N = 4096; biasRow = true;
  }

  int gw = blockIdx.x * 8 + wid;  // 0..1023
  int ctiles = N >> 6;
  int rt = gw / ctiles, ct = gw % ctiles;
  int row0 = rt * 16, col0 = ct * 64;

  int m = lane & 15, hi = (lane >> 4) & 1, n = lane & 15;
  const float* Abase = A + (size_t)(row0 + m) * 256 + hi * 16;
  const float* Bbase[4];
#pragma unroll
  for (int i = 0; i < 4; ++i)
    Bbase[i] = B + (size_t)(col0 + i * 16 + n) * 256 + hi * 16;

  v8f c[4];
#pragma unroll
  for (int i = 0; i < 4; ++i)
#pragma unroll
    for (int j = 0; j < 8; ++j) c[i][j] = 0.0f;

#pragma unroll
  for (int ks = 0; ks < 8; ++ks) {
    v16bf a = a_frag_f32_at(Abase, ks * 32, 1.0f);
#pragma unroll
    for (int i = 0; i < 4; ++i) {
      v16bf b = b_frag_f32_at(Bbase[i], ks * 32);
      c[i] = wmma_bf16(a, b, c[i]);
    }
  }

#pragma unroll
  for (int i = 0; i < 4; ++i) {
#pragma unroll
    for (int j = 0; j < 8; ++j) {
      int row = row0 + j + 8 * hi;
      int col = col0 + i * 16 + n;
      float v = c[i][j] + (biasRow ? bias[row] : bias[col]);
      C[(size_t)row * N + col] = (__bf16)v;
    }
  }
}

// ===========================================================================
// Phase 2: flash-style attention with exp(sigmoid(s/16)) weights (bounded in
// (1,e) -> no max tracking). grid = (64 row-groups, cnt blocks), block = 256.
// Wave pair (tile) shares 16 query rows: half=0/1 computes one 16-key subtile
// of S and one 128-wide half of O. Row-sums via one WMMA against ones-B.
// K/Vt chunks double-buffered in LDS via async-to-LDS loads (ASYNCcnt); the
// P-staging buffer aliases the already-consumed K buffer to fit 64 KB.
// Output: acc[n,d] += P@V / rowsum (atomic f32, mean folded later).
// ===========================================================================
__global__ __launch_bounds__(256) void attn_kernel(
    const __bf16* __restrict__ Qg, const __bf16* __restrict__ Kg,
    const __bf16* __restrict__ Vtg, float* __restrict__ acc) {
  __shared__ __bf16 KsL[2][32 * 256];  // 2 x 16 KB key chunks [32,256]
  __shared__ __bf16 VtL[2][256 * 32];  // 2 x 16 KB Vt chunks [256,32]

  int tid = threadIdx.x, lane = tid & 31, wid = tid >> 5;
  int tile = wid >> 1, half = wid & 1;
  int slot = blockIdx.y;
  const __bf16* Q = Qg + (size_t)slot * 1048576;
  const __bf16* K = Kg + (size_t)slot * 1048576;
  const __bf16* Vt = Vtg + (size_t)slot * 1048576;
  int row0 = blockIdx.x * 64 + tile * 16;
  int m = lane & 15, hi = (lane >> 4) & 1, n = lane & 15;

  // Stage one 32-key chunk (K: contiguous 16 KB; Vt: 256 rows x 64 B).
  auto stage = [&](int kb, int buf) {
    const uint4* src = (const uint4*)(K + (size_t)kb * 8192);
    uint4* dst = (uint4*)&KsL[buf][0];
    const uint4* vsrc = (const uint4*)(Vt + (size_t)tid * 4096 + kb * 32);
    uint4* vdst = (uint4*)(&VtL[buf][0] + tid * 32);
#if USE_ASYNC
#pragma unroll
    for (int i = 0; i < 4; ++i) async_b128(src + tid + i * 256, dst + tid + i * 256);
#pragma unroll
    for (int i = 0; i < 4; ++i) async_b128(vsrc + i, vdst + i);
#else
#pragma unroll
    for (int i = 0; i < 4; ++i) dst[tid + i * 256] = src[tid + i * 256];
#pragma unroll
    for (int i = 0; i < 4; ++i) vdst[i] = vsrc[i];
    if (kb + 1 < 128) {  // global_prefetch_b8 for the next chunk
      __builtin_prefetch(K + (size_t)(kb + 1) * 8192 + tid * 32, 0, 0);
      __builtin_prefetch(Vt + (size_t)tid * 4096 + (kb + 1) * 32, 0, 0);
    }
#endif
  };

  // Query tile resident in registers as 8 ready-made A fragments (K=256).
  const __bf16* Qbase = Q + (size_t)(row0 + m) * 256 + hi * 16;
  v16bf aq[8];
#pragma unroll
  for (int ks = 0; ks < 8; ++ks) aq[ks] = a_frag_bf_at(Qbase, ks * 32);

  v8f o[8];
#pragma unroll
  for (int d = 0; d < 8; ++d)
#pragma unroll
    for (int j = 0; j < 8; ++j) o[d][j] = 0.0f;
  v8f crs;
#pragma unroll
  for (int j = 0; j < 8; ++j) crs[j] = 0.0f;

  v16bf ones;
#pragma unroll
  for (int i = 0; i < 16; ++i) ones[i] = (__bf16)1.0f;

  // Prologue: stage chunk 0 into buffer 0.
  stage(0, 0);
#if USE_ASYNC
  wait_async0();
#endif
  __syncthreads();

  for (int kb = 0; kb < 128; ++kb) {
    int p = kb & 1;
    if (kb + 1 < 128) stage(kb + 1, p ^ 1);  // overlap with compute

    // S subtile: 16 query rows x 16 keys (this wave's half), K-dim 256.
    const __bf16* KsBase = &KsL[p][0] + (size_t)(half * 16 + n) * 256 + hi * 16;
    v8f c;
#pragma unroll
    for (int j = 0; j < 8; ++j) c[j] = 0.0f;
#pragma unroll
    for (int ks = 0; ks < 8; ++ks) {
      v16bf b = b_frag_bf_at(KsBase, ks * 32);
      c = wmma_bf16(aq[ks], b, c);
    }
    __syncthreads();  // all KsL[p] reads done -> safe to alias P over it

    // f = exp(sigmoid(s/16)); write bf16 P tile into the dead K buffer.
    __bf16* Ss = &KsL[p][0] + tile * 512;  // [16,32] per wave pair
#pragma unroll
    for (int j = 0; j < 8; ++j) {
      float s = c[j];
      float e = __builtin_amdgcn_exp2f(-s * (ATTN_SCALE * LOG2E));
      float sg = __builtin_amdgcn_rcpf(1.0f + e);
      float f = __builtin_amdgcn_exp2f(sg * LOG2E);
      Ss[(j + 8 * hi) * 32 + half * 16 + n] = (__bf16)f;
    }
    __syncthreads();  // partner's half of P visible

    // P as A fragment (16x32 over this key chunk), row-sum + O accumulation.
    v16bf ap = a_frag_bf_at(Ss + m * 32 + hi * 16, 0);
    crs = wmma_bf16(ap, ones, crs);  // rowsum replicated in every column
    const __bf16* VtBase = &VtL[p][0] + (size_t)(half * 128 + n) * 32 + hi * 16;
#pragma unroll
    for (int d = 0; d < 8; ++d) {
      v16bf b = b_frag_bf_at(VtBase, d * 512);
      o[d] = wmma_bf16(ap, b, o[d]);
    }

#if USE_ASYNC
    wait_async0();  // next buffer's async loads landed
#endif
    __syncthreads();  // everyone done with buffer p; swap
  }

  // Normalize and accumulate into the global block-sum.
#pragma unroll
  for (int j = 0; j < 8; ++j) {
    float inv = __builtin_amdgcn_rcpf(crs[j]);
    int row = row0 + j + 8 * hi;
#pragma unroll
    for (int d = 0; d < 8; ++d) {
      int col = half * 128 + d * 16 + n;
      atomicAdd(&acc[(size_t)row * 256 + col], o[d][j] * inv);
    }
  }
}

// ===========================================================================
// Phase 3: MLP layers, f32 in/out, bf16 WMMA. mode 0: relu(A@B^T*ascale+bias);
// mode 1: sigmoid(A@B^T) + bias (final layer; B is pre-transposed weight).
// grid = 128, block = 256 (M=4096, N=256 fixed).
// ===========================================================================
__global__ __launch_bounds__(256) void mlp_gemm(
    const float* __restrict__ A, const float* __restrict__ B,
    const float* __restrict__ bias, float* __restrict__ C, float ascale,
    int mode) {
  int tid = threadIdx.x, lane = tid & 31, wid = tid >> 5;
  int gw = blockIdx.x * 8 + wid;
  int rt = gw >> 2, ct = gw & 3;
  int row0 = rt * 16, col0 = ct * 64;
  int m = lane & 15, hi = (lane >> 4) & 1, n = lane & 15;

  const float* Abase = A + (size_t)(row0 + m) * 256 + hi * 16;
  const float* Bbase[4];
#pragma unroll
  for (int i = 0; i < 4; ++i)
    Bbase[i] = B + (size_t)(col0 + i * 16 + n) * 256 + hi * 16;

  v8f c[4];
#pragma unroll
  for (int i = 0; i < 4; ++i)
#pragma unroll
    for (int j = 0; j < 8; ++j) c[i][j] = 0.0f;

#pragma unroll
  for (int ks = 0; ks < 8; ++ks) {
    v16bf a = a_frag_f32_at(Abase, ks * 32, ascale);
#pragma unroll
    for (int i = 0; i < 4; ++i) {
      v16bf b = b_frag_f32_at(Bbase[i], ks * 32);
      c[i] = wmma_bf16(a, b, c[i]);
    }
  }

#pragma unroll
  for (int i = 0; i < 4; ++i) {
#pragma unroll
    for (int j = 0; j < 8; ++j) {
      int row = row0 + j + 8 * hi;
      int col = col0 + i * 16 + n;
      float v = c[i][j];
      if (mode == 0) {
        v = fmaxf(v + bias[col], 0.0f);
      } else {
        v = __builtin_amdgcn_rcpf(1.0f + __builtin_amdgcn_exp2f(-v * LOG2E)) +
            bias[col];
      }
      C[(size_t)row * 256 + col] = v;
    }
  }
}

__global__ void transpose256(const float* __restrict__ in,
                             float* __restrict__ out) {
  int r = blockIdx.x, c = threadIdx.x;
  out[(size_t)c * 256 + r] = in[(size_t)r * 256 + c];
}

// ===========================================================================
// Host launcher
// ===========================================================================
extern "C" void kernel_launch(void* const* d_in, const int* in_sizes, int n_in,
                              void* d_out, int out_size, void* d_ws,
                              size_t ws_size, hipStream_t stream) {
  (void)in_sizes; (void)n_in; (void)out_size;
  const float* x  = (const float*)d_in[0];
  const float* Wq = (const float*)d_in[1];
  const float* bq = (const float*)d_in[2];
  const float* Wk = (const float*)d_in[3];
  const float* bk = (const float*)d_in[4];
  const float* Wv = (const float*)d_in[5];
  const float* bv = (const float*)d_in[6];
  const float* Wl = (const float*)d_in[7];
  const float* bl = (const float*)d_in[8];
  const float* Fw = (const float*)d_in[9];
  const float* Fb = (const float*)d_in[10];
  float* out = (float*)d_out;

  // Workspace layout: acc (4MB) | h0 (4MB) | h1 (4MB) | FwT (256KB) | bf16 slabs
  float* acc = (float*)d_ws;
  float* h0  = acc + 1048576;
  float* h1  = h0 + 1048576;
  float* FwT = h1 + 1048576;
  __bf16* slabs = (__bf16*)(FwT + 65536);
  size_t base_bytes = (size_t)(3 * 1048576 + 65536) * 4;

  int G = 1;
  if (ws_size > base_bytes) {
    size_t g = (ws_size - base_bytes) / ((size_t)3 * 1048576 * 2);
    G = (g > 8) ? 8 : (g < 1 ? 1 : (int)g);
  }
  __bf16* Qslot  = slabs;
  __bf16* Kslot  = Qslot + (size_t)G * 1048576;
  __bf16* Vtslot = Kslot + (size_t)G * 1048576;

  (void)hipMemsetAsync(acc, 0, (size_t)1048576 * 4, stream);

  for (int b0 = 0; b0 < 200; b0 += G) {
    int cnt = (200 - b0 < G) ? (200 - b0) : G;
    qkv_gemm<<<dim3(128, 3, cnt), 256, 0, stream>>>(x, Wq, Wk, Wv, bq, bk, bv,
                                                    Qslot, Kslot, Vtslot, b0);
    attn_kernel<<<dim3(64, cnt), 256, 0, stream>>>(Qslot, Kslot, Vtslot, acc);
  }

  // MLP: first layer folds the mean (1/200).
  const float* hin = acc;
  float* hout = h0;
  float sc = 1.0f / 200.0f;
  for (int l = 0; l < 20; ++l) {
    mlp_gemm<<<128, 256, 0, stream>>>(hin, Wl + (size_t)l * 65536,
                                      bl + (size_t)l * 256, hout, sc, 0);
    hin = hout;
    hout = (hout == h0) ? h1 : h0;
    sc = 1.0f;
  }
  transpose256<<<256, 256, 0, stream>>>(Fw, FwT);
  mlp_gemm<<<128, 256, 0, stream>>>(hin, FwT, Fb, out, 1.0f, 1);
}
